// MultiHeadAttention_77360950936212
// MI455X (gfx1250) — matrix operand order repaired and checked by reading, not compiled, not run on hardware
//
#include <hip/hip_runtime.h>

typedef __bf16 bf16;
typedef __attribute__((ext_vector_type(16))) __bf16 v16bf;
typedef __attribute__((ext_vector_type(8)))  __bf16 v8bf;
typedef __attribute__((ext_vector_type(8)))  float  v8f;
typedef __attribute__((ext_vector_type(4)))  unsigned int u32x4;
typedef __attribute__((ext_vector_type(8)))  int i32x8;
typedef __attribute__((ext_vector_type(4)))  int i32x4;

#define S_LEN  2048
#define D_DIM  1024
#define NH     16
#define DKH    64
#define BATCH  4
#define MROWS  (BATCH * S_LEN)   // 8192

#if __has_builtin(__builtin_amdgcn_tensor_load_to_lds) && __has_builtin(__builtin_amdgcn_s_wait_tensorcnt)
#define HAVE_TDM 1
#else
#define HAVE_TDM 0
#endif

// ---------- helpers ----------
__device__ __forceinline__ bf16 f2bf(float f) { return (bf16)f; }

__device__ __forceinline__ v8bf cvt8(float4 a, float4 b) {
  v8bf r;
  r[0] = (bf16)a.x; r[1] = (bf16)a.y; r[2] = (bf16)a.z; r[3] = (bf16)a.w;
  r[4] = (bf16)b.x; r[5] = (bf16)b.y; r[6] = (bf16)b.z; r[7] = (bf16)b.w;
  return r;
}
__device__ __forceinline__ v8f vzero8() {
  v8f r;
#pragma unroll
  for (int i = 0; i < 8; ++i) r[i] = 0.f;
  return r;
}
__device__ __forceinline__ v16bf combine8(v8bf x, v8bf y) {
  v16bf r;
#pragma unroll
  for (int i = 0; i < 8; ++i) { r[i] = x[i]; r[i + 8] = y[i]; }
  return r;
}
// A fragment (16x32 bf16): lane(lo,hi) holds M=lo, K = {g*16 + hi*8 + 0..7 | g=0,1}
__device__ __forceinline__ v16bf frag_a(const bf16* rowk, int hi) {
  v8bf x = *(const v8bf*)(rowk + hi * 8);
  v8bf y = *(const v8bf*)(rowk + 16 + hi * 8);
  return combine8(x, y);
}
// B fragment (32x16 bf16, tile stored [N][K]): lane(lo,hi) holds N=lo, K = {g*16 + hi*8 + 0..7 | g=0,1}
__device__ __forceinline__ v16bf frag_b(const bf16* rowk, int hi) {
  v8bf x = *(const v8bf*)(rowk + hi * 8);
  v8bf y = *(const v8bf*)(rowk + 16 + hi * 8);
  return combine8(x, y);
}

#if HAVE_TDM
// TDM 2-D tile load: bf16 elements, row padding in LDS expressed via D# pad fields.
// D# layout per CDNA5 ISA (count=1, type=2, data_size=1 -> 2B).
__device__ __forceinline__ void tdm_load_2d(unsigned lds_off, const void* gaddr,
                                            unsigned tile_d0, unsigned tile_d1,
                                            unsigned stride0,
                                            unsigned pad_interval_code,
                                            unsigned pad_amount_code) {
  unsigned long long ga = (unsigned long long)gaddr;
  u32x4 g0;
  g0[0] = 1u;                                   // count=1, is_restore=0, gather=0
  g0[1] = lds_off;                              // LDS byte offset (wave-relative)
  g0[2] = (unsigned)ga;                         // global_addr[31:0]
  g0[3] = ((unsigned)(ga >> 32) & 0x01FFFFFFu) | (2u << 30);  // addr[56:32] | type=2

  unsigned w0 = (1u << 16);                     // data_size = 2 bytes
  w0 |= (1u << 20) | (pad_interval_code << 22) | (pad_amount_code << 25); // pad_enable
  i32x8 g1;
  g1[0] = (int)w0;
  g1[1] = (int)((tile_d0 & 0xFFFFu) << 16);               // tensor_dim0[15:0] (== tile_d0)
  g1[2] = (int)((tile_d0 >> 16) | ((tile_d1 & 0xFFFFu) << 16));  // dim0 hi | tensor_dim1 lo
  g1[3] = (int)((tile_d1 >> 16) | (tile_d0 << 16));       // dim1 hi | tile_dim0
  g1[4] = (int)(tile_d1 & 0xFFFFu);                       // tile_dim1 (tile_dim2 = 0)
  g1[5] = (int)stride0;                                   // tensor_dim0_stride[31:0]
  g1[6] = 0;                                              // stride0 hi | dim1_stride lo
  g1[7] = 0;
  i32x4 g2 = {0, 0, 0, 0};
  i32x4 g3 = {0, 0, 0, 0};
#if __clang_major__ >= 23
  i32x8 g4 = {0, 0, 0, 0, 0, 0, 0, 0};
  __builtin_amdgcn_tensor_load_to_lds(g0, g1, g2, g3, g4, 0);
#else
  __builtin_amdgcn_tensor_load_to_lds(g0, g1, g2, g3, 0);
#endif
}
__device__ __forceinline__ unsigned lds_offset_of(const void* p) {
  return (unsigned)(unsigned long long)(uintptr_t)p;  // generic LDS ptr: addr[31:0] = LDS offset
}
#endif

// ---------- kernel 1: fused QKV projection  Y = X @ W^T + b -> [B,H,S,DK] bf16 ----------
#define PBM 128
#define PBN 128
#define LDT 40   // 32 + 8 bf16 padding (80B rows: 16B aligned, bank-spread)

__global__ __launch_bounds__(256) void qkv_proj_kernel(
    const float* __restrict__ xq, const float* __restrict__ xk, const float* __restrict__ xv,
    const float* __restrict__ Wq, const float* __restrict__ bq,
    const float* __restrict__ Wk, const float* __restrict__ bk,
    const float* __restrict__ Wv, const float* __restrict__ bv,
    unsigned short* __restrict__ qh_us, unsigned short* __restrict__ kh_us,
    unsigned short* __restrict__ vh_us)
{
  __shared__ bf16 sA[PBM * LDT];
  __shared__ bf16 sB[PBN * LDT];

  const int which = blockIdx.z;
  const float* X    = (which == 0) ? xq : (which == 1) ? xk : xv;
  const float* W    = (which == 0) ? Wq : (which == 1) ? Wk : Wv;
  const float* bias = (which == 0) ? bq : (which == 1) ? bk : bv;
  bf16* out = (bf16*)((which == 0) ? qh_us : (which == 1) ? kh_us : vh_us);

  const int tid = threadIdx.x;
  const int lo = tid & 15, hi = (tid >> 4) & 1;
  const int w  = tid >> 5;
  const int wm = w >> 1;        // 0..3 -> 32-row slabs
  const int wn = w & 1;         // 0..1 -> 64-col slabs
  const int m0 = blockIdx.x * PBM;
  const int n0 = blockIdx.y * PBN;

  v8f acc[2][4];
#pragma unroll
  for (int i = 0; i < 2; ++i)
#pragma unroll
    for (int j = 0; j < 4; ++j) acc[i][j] = vzero8();

  const int lrow = tid >> 1;
  const int lcol = (tid & 1) * 16;

  for (int k0 = 0; k0 < D_DIM; k0 += 32) {
    __syncthreads();
    {
      const float4* src = (const float4*)(X + (size_t)(m0 + lrow) * D_DIM + k0 + lcol);
      float4 f0 = src[0], f1 = src[1], f2 = src[2], f3 = src[3];
      *(v8bf*)&sA[lrow * LDT + lcol]     = cvt8(f0, f1);
      *(v8bf*)&sA[lrow * LDT + lcol + 8] = cvt8(f2, f3);
    }
    {
      const float4* src = (const float4*)(W + (size_t)(n0 + lrow) * D_DIM + k0 + lcol);
      float4 f0 = src[0], f1 = src[1], f2 = src[2], f3 = src[3];
      *(v8bf*)&sB[lrow * LDT + lcol]     = cvt8(f0, f1);
      *(v8bf*)&sB[lrow * LDT + lcol + 8] = cvt8(f2, f3);
    }
    __syncthreads();

    v16bf afr[2];
#pragma unroll
    for (int mi = 0; mi < 2; ++mi)
      afr[mi] = frag_a(&sA[(wm * 32 + mi * 16 + lo) * LDT], hi);
#pragma unroll
    for (int ni = 0; ni < 4; ++ni) {
      v16bf bfr = frag_b(&sB[(wn * 64 + ni * 16 + lo) * LDT], hi);
#pragma unroll
      for (int mi = 0; mi < 2; ++mi)
        acc[mi][ni] = __builtin_amdgcn_wmma_f32_16x16x32_bf16(
            false, afr[mi], false, bfr, (short)0, acc[mi][ni], false, false);
    }
  }

  // epilogue: bias, convert, scatter to [B,H,S,DK]
#pragma unroll
  for (int ni = 0; ni < 4; ++ni) {
    const int n = n0 + wn * 64 + ni * 16 + lo;
    const float bb = bias[n];
    const int hh = n >> 6, dk = n & 63;
#pragma unroll
    for (int mi = 0; mi < 2; ++mi) {
#pragma unroll
      for (int i = 0; i < 8; ++i) {
        const int m = m0 + wm * 32 + mi * 16 + i + hi * 8;
        const int bidx = m >> 11, s = m & (S_LEN - 1);
        out[(((size_t)bidx * NH + hh) * S_LEN + s) * DKH + dk] = f2bf(acc[mi][ni][i] + bb);
      }
    }
  }
}

// ---------- kernel 2: flash attention (online softmax) ----------
#define QTILE 128
#define LQ 72
#define LK 72
#define LV 40
#define LP 40

__global__ __launch_bounds__(256) void flash_attn_kernel(
    const unsigned short* __restrict__ qh_us, const unsigned short* __restrict__ kh_us,
    const unsigned short* __restrict__ vh_us, const int* __restrict__ mask,
    unsigned short* __restrict__ ctx_us)
{
  __shared__ bf16 sQ[QTILE * LQ];   // [128][64] + pad
  __shared__ bf16 sK[32 * LK];      // [32 keys][64 dk] + pad
  __shared__ bf16 sVt[64 * LV];     // transposed: [64 dk][32 keys] + pad
  __shared__ bf16 sP[8 * 16 * LP];  // per-wave P staging [16][32] + pad

  const int tid = threadIdx.x;
  const int lo = tid & 15, hi = (tid >> 4) & 1;
  const int w  = tid >> 5;
  const int bh = blockIdx.y;
  const int b  = bh >> 4, h = bh & 15;
  const int q0 = blockIdx.x * QTILE;

  const bf16* Q = (const bf16*)qh_us + (size_t)bh * S_LEN * DKH;
  const bf16* K = (const bf16*)kh_us + (size_t)bh * S_LEN * DKH;
  const bf16* V = (const bf16*)vh_us + (size_t)bh * S_LEN * DKH;
  bf16* ctx = (bf16*)ctx_us;

#if HAVE_TDM
  // Q tile 128x64 bf16 via Tensor Data Mover; LDS padding 4 DWORDs per 32 DWORDs -> LQ=72
  if (w == 0) {
    tdm_load_2d(lds_offset_of(&sQ[0]), Q + (size_t)q0 * DKH,
                /*tile_d0=*/DKH, /*tile_d1=*/QTILE, /*stride0=*/DKH,
                /*pad_interval_code(32dw)=*/4, /*pad_amount_code(4dw)=*/3);
    __builtin_amdgcn_s_wait_tensorcnt(0);
  }
#else
  {  // Q tile -> LDS (bf16 straight copy)
    const int row = tid >> 1;
    const int c0 = (tid & 1) * 32;
    const uint4* src = (const uint4*)(Q + (size_t)(q0 + row) * DKH + c0);
#pragma unroll
    for (int j = 0; j < 4; ++j)
      *(uint4*)&sQ[row * LQ + c0 + j * 8] = src[j];
  }
#endif

  float mstat[8], lstat[8];
  v8f acc[4];
#pragma unroll
  for (int i = 0; i < 8; ++i) { mstat[i] = -3.0e38f; lstat[i] = 0.f; }
#pragma unroll
  for (int n = 0; n < 4; ++n) acc[n] = vzero8();

  const int r0 = w * 16;  // this wave's 16 q-rows within the tile

  for (int kt = 0; kt < S_LEN; kt += 32) {
    __syncthreads();
#if HAVE_TDM
    if (w == 0) {  // K tile 32x64 via TDM, overlapped with the manual V transpose below
      tdm_load_2d(lds_offset_of(&sK[0]), K + (size_t)kt * DKH,
                  DKH, 32, DKH, 4, 3);
    }
#else
    {  // K tile: [32][64]
      const int row = tid >> 3;
      const int c0 = (tid & 7) * 8;
      *(uint4*)&sK[row * LK + c0] = *(const uint4*)(K + (size_t)(kt + row) * DKH + c0);
    }
#endif
    {  // V tile transposed into [dk][key]
      const int row = tid >> 3;       // key 0..31
      const int c0 = (tid & 7) * 8;   // dk base
      v8bf vv = *(const v8bf*)(V + (size_t)(kt + row) * DKH + c0);
#pragma unroll
      for (int j = 0; j < 8; ++j)
        sVt[(c0 + j) * LV + row] = vv[j];
    }
    if (kt + 32 < S_LEN)  // prefetch next V tile into GL2 (global_prefetch_b8)
      __builtin_prefetch(V + (size_t)(kt + 32) * DKH + tid * 16, 0, 1);
#if HAVE_TDM
    if (w == 0) __builtin_amdgcn_s_wait_tensorcnt(0);
#endif
    __syncthreads();

    // scores S = Q @ K^T : 16 rows x 32 keys per wave
    v8f s0 = vzero8(), s1 = vzero8();
#pragma unroll
    for (int kk = 0; kk < 2; ++kk) {
      v16bf a  = frag_a(&sQ[(r0 + lo) * LQ + kk * 32], hi);
      v16bf b0 = frag_b(&sK[lo * LK + kk * 32], hi);
      v16bf b1 = frag_b(&sK[(16 + lo) * LK + kk * 32], hi);
      s0 = __builtin_amdgcn_wmma_f32_16x16x32_bf16(false, a, false, b0, (short)0, s0, false, false);
      s1 = __builtin_amdgcn_wmma_f32_16x16x32_bf16(false, a, false, b1, (short)0, s1, false, false);
    }

    // scale + mask + online softmax (row = i + hi*8, col = lo / 16+lo)
#pragma unroll
    for (int i = 0; i < 8; ++i) {
      const int qrow = q0 + r0 + i + hi * 8;
      float v0 = s0[i] * 0.125f;   // 1/sqrt(64)
      float v1 = s1[i] * 0.125f;
      const int mm0 = mask[(size_t)qrow * S_LEN + kt + lo];
      const int mm1 = mask[(size_t)qrow * S_LEN + kt + 16 + lo];
      v0 = mm0 ? v0 : -1.0e9f;
      v1 = mm1 ? v1 : -1.0e9f;
      float t = fmaxf(v0, v1);
      t = fmaxf(t, __shfl_xor(t, 1, 32));
      t = fmaxf(t, __shfl_xor(t, 2, 32));
      t = fmaxf(t, __shfl_xor(t, 4, 32));
      t = fmaxf(t, __shfl_xor(t, 8, 32));
      const float mnew  = fmaxf(mstat[i], t);
      const float alpha = __expf(mstat[i] - mnew);
      const float p0 = __expf(v0 - mnew);
      const float p1 = __expf(v1 - mnew);
      float rs = p0 + p1;
      rs += __shfl_xor(rs, 1, 32);
      rs += __shfl_xor(rs, 2, 32);
      rs += __shfl_xor(rs, 4, 32);
      rs += __shfl_xor(rs, 8, 32);
      lstat[i] = lstat[i] * alpha + rs;
      mstat[i] = mnew;
#pragma unroll
      for (int n = 0; n < 4; ++n) acc[n][i] *= alpha;
      // stage P into wave-private LDS for A-fragment re-layout
      sP[(w * 16 + i + hi * 8) * LP + lo]      = f2bf(p0);
      sP[(w * 16 + i + hi * 8) * LP + 16 + lo] = f2bf(p1);
    }

    // acc += P @ V  (contraction over 32 keys, N = dk 0..63)
    v16bf ap = frag_a(&sP[(w * 16 + lo) * LP], hi);
#pragma unroll
    for (int n = 0; n < 4; ++n) {
      v16bf bv = frag_b(&sVt[(n * 16 + lo) * LV], hi);
      acc[n] = __builtin_amdgcn_wmma_f32_16x16x32_bf16(false, ap, false, bv, (short)0, acc[n], false, false);
    }
  }

  // normalize and store context as [B,S,H*DK] bf16
#pragma unroll
  for (int i = 0; i < 8; ++i) {
    const int qrow = q0 + r0 + i + hi * 8;
    const float inv = 1.0f / lstat[i];
#pragma unroll
    for (int n = 0; n < 4; ++n)
      ctx[((size_t)b * S_LEN + qrow) * D_DIM + h * DKH + n * 16 + lo] = f2bf(acc[n][i] * inv);
  }
}

// ---------- kernel 3: output projection  out = C @ Wo^T + bo (fp32 out) ----------
__global__ __launch_bounds__(256) void out_proj_kernel(
    const unsigned short* __restrict__ ctx_us, const float* __restrict__ Wo,
    const float* __restrict__ bo, float* __restrict__ out)
{
  __shared__ bf16 sA[PBM * LDT];
  __shared__ bf16 sB[PBN * LDT];

  const bf16* C = (const bf16*)ctx_us;
  const int tid = threadIdx.x;
  const int lo = tid & 15, hi = (tid >> 4) & 1;
  const int w  = tid >> 5;
  const int wm = w >> 1;
  const int wn = w & 1;
  const int m0 = blockIdx.x * PBM;
  const int n0 = blockIdx.y * PBN;

  v8f acc[2][4];
#pragma unroll
  for (int i = 0; i < 2; ++i)
#pragma unroll
    for (int j = 0; j < 4; ++j) acc[i][j] = vzero8();

  const int lrow = tid >> 1;
  const int lcol = (tid & 1) * 16;

  for (int k0 = 0; k0 < D_DIM; k0 += 32) {
    __syncthreads();
    {  // A: bf16 copy
      const uint4* src = (const uint4*)(C + (size_t)(m0 + lrow) * D_DIM + k0 + lcol);
      *(uint4*)&sA[lrow * LDT + lcol]     = src[0];
      *(uint4*)&sA[lrow * LDT + lcol + 8] = src[1];
    }
    {  // B: fp32 -> bf16
      const float4* src = (const float4*)(Wo + (size_t)(n0 + lrow) * D_DIM + k0 + lcol);
      float4 f0 = src[0], f1 = src[1], f2 = src[2], f3 = src[3];
      *(v8bf*)&sB[lrow * LDT + lcol]     = cvt8(f0, f1);
      *(v8bf*)&sB[lrow * LDT + lcol + 8] = cvt8(f2, f3);
    }
    __syncthreads();

    v16bf afr[2];
#pragma unroll
    for (int mi = 0; mi < 2; ++mi)
      afr[mi] = frag_a(&sA[(wm * 32 + mi * 16 + lo) * LDT], hi);
#pragma unroll
    for (int ni = 0; ni < 4; ++ni) {
      v16bf bfr = frag_b(&sB[(wn * 64 + ni * 16 + lo) * LDT], hi);
#pragma unroll
      for (int mi = 0; mi < 2; ++mi)
        acc[mi][ni] = __builtin_amdgcn_wmma_f32_16x16x32_bf16(
            false, afr[mi], false, bfr, (short)0, acc[mi][ni], false, false);
    }
  }

#pragma unroll
  for (int ni = 0; ni < 4; ++ni) {
    const int n = n0 + wn * 64 + ni * 16 + lo;
    const float bb = bo[n];
#pragma unroll
    for (int mi = 0; mi < 2; ++mi) {
#pragma unroll
      for (int i = 0; i < 8; ++i) {
        const int m = m0 + wm * 32 + mi * 16 + i + hi * 8;
        out[(size_t)m * D_DIM + n] = acc[mi][ni][i] + bb;
      }
    }
  }
}

// ---------- launch ----------
extern "C" void kernel_launch(void* const* d_in, const int* in_sizes, int n_in,
                              void* d_out, int out_size, void* d_ws, size_t ws_size,
                              hipStream_t stream) {
  const float* q   = (const float*)d_in[0];
  const float* k   = (const float*)d_in[1];
  const float* v   = (const float*)d_in[2];
  const int*   msk = (const int*)d_in[3];
  const float* Wq  = (const float*)d_in[4];
  const float* bq  = (const float*)d_in[5];
  const float* Wk  = (const float*)d_in[6];
  const float* bk  = (const float*)d_in[7];
  const float* Wv  = (const float*)d_in[8];
  const float* bv  = (const float*)d_in[9];
  const float* Wo  = (const float*)d_in[10];
  const float* bo  = (const float*)d_in[11];
  float* out = (float*)d_out;

  const size_t HEAD_ELEMS = (size_t)BATCH * NH * S_LEN * DKH;  // 8,388,608
  unsigned short* qh  = (unsigned short*)d_ws;
  unsigned short* kh  = qh + HEAD_ELEMS;
  unsigned short* vh  = kh + HEAD_ELEMS;
  unsigned short* ctx = vh + HEAD_ELEMS;   // 64 MB total bf16 workspace

  dim3 pgrid(MROWS / PBM, D_DIM / PBN, 3);
  qkv_proj_kernel<<<pgrid, 256, 0, stream>>>(q, k, v, Wq, bq, Wk, bk, Wv, bv, qh, kh, vh);

  dim3 fgrid(S_LEN / QTILE, BATCH * NH);
  flash_attn_kernel<<<fgrid, 256, 0, stream>>>(qh, kh, vh, msk, ctx);

  dim3 ogrid(MROWS / PBM, D_DIM / PBN);
  out_proj_kernel<<<ogrid, 256, 0, stream>>>(ctx, Wo, bo, out);
}
